// CrossAttention_10857677324980
// MI455X (gfx1250) — matrix operand order repaired and checked by reading, not compiled, not run on hardware
//
#include <hip/hip_runtime.h>
#include <hip/hip_fp16.h>
#include <math.h>

typedef __attribute__((ext_vector_type(16))) _Float16 v16h;
typedef __attribute__((ext_vector_type(8)))  _Float16 v8h;
typedef __attribute__((ext_vector_type(4)))  _Float16 v4h;
typedef __attribute__((ext_vector_type(8)))  float    v8f;

#define N_CAM 6
#define N_PTS 4
#define HID   256
#define CFE   256
#define HF    28
#define WF    50
#define KDIM  256          // K of both GEMMs (hidden size)
#define EPSV  1e-5f
#define PI_F  3.14159265358979323846f

// ---------------- prep: q16[(b*nq+q)*HID+c] = f16(query + query_pos); b = blockIdx.y
__global__ void k_prep_q16(const float* __restrict__ query,
                           const float* __restrict__ qpos,
                           _Float16* __restrict__ q16,
                           int nq, int bs) {
  int b = blockIdx.y;
  int d = blockIdx.x * blockDim.x + threadIdx.x;   // over nq*HID/4
  int c4 = (d << 2) & (HID - 1);
  int q  = (d << 2) >> 8;                           // HID == 256
  if (q >= nq) return;
  size_t src = ((size_t)q * bs + b) * HID + c4;
  const float4 a = *(const float4*)(query + src);
  const float4 p = *(const float4*)(qpos + src);
  v4h o = { (_Float16)(a.x + p.x), (_Float16)(a.y + p.y),
            (_Float16)(a.z + p.z), (_Float16)(a.w + p.w) };
  *(v4h*)(q16 + ((size_t)b * nq + q) * HID + c4) = o;
}

// ---------------- prep: B[K=256, cin] f32 -> fragment-native swizzled f16
// dst layout: [ntile][kt][lane][16e], so gemm B-frag = one contiguous 32B load per lane.
// Fragment map (ISA 7.12.2, 16-bit B 32x16): n = ntile*16 + lane%16,
//   k = kt*32 + 16*(e/8) + 8*(lane/16) + (e&7)
__global__ void k_prep_bswz(const float* __restrict__ src, _Float16* __restrict__ dst,
                            int N, int cin) {
  int d = blockIdx.x * blockDim.x + threadIdx.x;
  if (d >= N * KDIM) return;
  int e     = d & 15;
  int lane  = (d >> 4) & 31;
  int kt    = (d >> 9) & 7;            // KDIM/32 == 8
  int ntile = (d >> 9) >> 3;
  int k = kt * 32 + ((e >> 3) << 4) + ((lane >> 4) << 3) + (e & 7);
  int n = (ntile << 4) + (lane & 15);
  dst[d] = (_Float16)((n < cin) ? src[k * cin + n] : 0.0f);
}

// ---------------- feats transpose: [b,n,C,H,W] -> [b,n,H,W,C] (channel-last)
__global__ void k_feats_T(const float* __restrict__ src, float* __restrict__ dst, int total) {
  int d = blockIdx.x * blockDim.x + threadIdx.x;
  if (d >= total) return;
  int c = d & (CFE - 1);
  int rest = d >> 8;
  int x = rest % WF; rest /= WF;
  int y = rest % HF; rest /= HF;       // rest = b*N_CAM + n
  dst[d] = src[((rest * CFE + c) * HF + y) * WF + x];
}

// ---------------- wave32 WMMA f16 GEMM, K=256 fully unrolled (8 x v_wmma)
// MODE 0: C[M,N] f32.  MODE 1: out[(q*bs+b)*N+col] = acc + bias[col] + resid[...]
template <int MODE>
__global__ void k_wmma_gemm(const _Float16* __restrict__ A, const _Float16* __restrict__ Bsw,
                            float* __restrict__ C, int M, int N,
                            const float* __restrict__ bias,
                            const float* __restrict__ resid, float* __restrict__ outp,
                            int nq, int bs) {
  int lane = threadIdx.x & 31;
  int wid  = (blockIdx.x * blockDim.x + threadIdx.x) >> 5;
  int Mt = M >> 4, Nt = N >> 4;
  if (wid >= Mt * Nt) return;
  int mtile = wid / Nt, ntile = wid % Nt;
  int row0 = mtile << 4, col0 = ntile << 4;
  int lm = lane & 15, half = lane >> 4;

  // A row pointer for this lane: 16B-aligned; each k-step is two contiguous b128 loads
  const _Float16* Ar = A + (size_t)(row0 + lm) * KDIM + (half << 3);
  const v16h* Bw = (const v16h*)(Bsw) + ((size_t)ntile << 8) + ((size_t)lane);  // [ntile][kt][lane]

  v8f acc = {};
#pragma unroll
  for (int kt = 0; kt < KDIM / 32; ++kt) {
    v8h alo = *(const v8h*)(Ar + kt * 32);
    v8h ahi = *(const v8h*)(Ar + kt * 32 + 16);
    v16h a  = __builtin_shufflevector(alo, ahi, 0, 1, 2, 3, 4, 5, 6, 7,
                                      8, 9, 10, 11, 12, 13, 14, 15);
    v16h bf = Bw[kt * 32];             // lane-contiguous 32B fragment
    acc = __builtin_amdgcn_wmma_f32_16x16x32_f16(false, a, false, bf,
                                                 (short)0, acc, false, false);
  }

  // D layout: VGPR r -> M = r + 8*half, N = lane%16
  if (MODE == 0) {
#pragma unroll
    for (int r = 0; r < 8; ++r) {
      int mm = row0 + r + 8 * half;
      C[(size_t)mm * N + col0 + lm] = acc[r];
    }
  } else {
    int b  = row0 / nq;                // one division per wave (nq % 16 == 0)
    int q0 = row0 - b * nq;
    int col = col0 + lm;
    float bv = bias[col];
#pragma unroll
    for (int r = 0; r < 8; ++r) {
      int q = q0 + r + 8 * half;
      size_t oi = ((size_t)q * bs + b) * N + col;
      outp[oi] = acc[r] + bv + resid[oi];
    }
  }
}

// ---------------- geometry: polar->cartesian, project, mask, sigmoid weights; b = blockIdx.y
__global__ void k_geom(const float* __restrict__ rho, const float* __restrict__ phi,
                       const float* __restrict__ zz, const float* __restrict__ l2i,
                       const float* __restrict__ intervals, const float* __restrict__ pcr,
                       const float* __restrict__ imsz,
                       const float* __restrict__ awbuf, const float* __restrict__ ba,
                       float* __restrict__ wbuf, float* __restrict__ ixy,
                       float* __restrict__ mask_out, int nq) {
  int b = blockIdx.y;
  int q = blockIdx.x * blockDim.x + threadIdx.x;
  if (q >= nq) return;
  int t = b * nq + q;

  float r = rho[q] * pcr[0] + intervals[0] * 0.5f;
  float p = phi[q] * pcr[1] - PI_F + intervals[1] * 0.5f;
  float xx = r * cosf(p);
  float yy = r * sinf(p);
  float zv = zz[t] * (pcr[3] - pcr[2]) + pcr[2];
  float iw = imsz[0], ih = imsz[1];

#pragma unroll
  for (int n = 0; n < N_CAM; ++n) {
    const float* Mx = l2i + (size_t)(b * N_CAM + n) * 16;
    float c0 = Mx[0] * xx + Mx[1] * yy + Mx[2] * zv + Mx[3];
    float c1 = Mx[4] * xx + Mx[5] * yy + Mx[6] * zv + Mx[7];
    float c2 = Mx[8] * xx + Mx[9] * yy + Mx[10] * zv + Mx[11];
    float den = fmaxf(c2, EPSV);
    float gx = c0 / den / iw;
    float gy = c1 / den / ih;
    float gxn = (gx - 0.5f) * 2.0f;
    float gyn = (gy - 0.5f) * 2.0f;
    bool msk = (c2 > EPSV) && (gxn > -1.0f) && (gxn < 1.0f) && (gyn > -1.0f) && (gyn < 1.0f);

    float w = 0.0f;
    if (msk) {
#pragma unroll
      for (int pp = 0; pp < N_PTS; ++pp) {
        float a = awbuf[(size_t)t * 32 + n * N_PTS + pp] + ba[n * N_PTS + pp];
        w += 1.0f / (1.0f + expf(-a));
      }
    }
    int mi = t * N_CAM + n;
    wbuf[mi] = w;
    ixy[mi * 2 + 0] = gx * (float)WF - 0.5f;    // align_corners=False pixel coords
    ixy[mi * 2 + 1] = gy * (float)HF - 0.5f;
    mask_out[mi] = msk ? 1.0f : 0.0f;
  }
}

// ---------------- bilinear sample + camera combine; block=(q), blockIdx.y=b,
// 64 threads x 4 channels, float4 corner loads (coalesced channel-last)
__global__ void k_sample(const float* __restrict__ featsT, const float* __restrict__ wbuf,
                         const float* __restrict__ ixy, _Float16* __restrict__ agg16,
                         int nq) {
  int b = blockIdx.y;
  int m = b * nq + blockIdx.x;
  int c4 = threadIdx.x << 2;
  __shared__ float sw[N_CAM], sx[N_CAM], sy[N_CAM];
  if (threadIdx.x < N_CAM) {
    int mi = m * N_CAM + threadIdx.x;
    sw[threadIdx.x] = wbuf[mi];
    sx[threadIdx.x] = ixy[mi * 2 + 0];
    sy[threadIdx.x] = ixy[mi * 2 + 1];
  }
  __syncthreads();

  float ax = 0.f, ay = 0.f, az = 0.f, aw = 0.f;
#pragma unroll
  for (int n = 0; n < N_CAM; ++n) {
    float w = sw[n];
    if (w == 0.0f) continue;            // masked camera, uniform across block
    float ix = sx[n], iy = sy[n];
    float x0f = floorf(ix), y0f = floorf(iy);
    float fx = ix - x0f, fy = iy - y0f;
    int x0 = (int)x0f, y0 = (int)y0f;
    const float* fb = featsT + (size_t)(b * N_CAM + n) * HF * WF * CFE + c4;

    float vx = 0.f, vy = 0.f, vz = 0.f, vw = 0.f;
#pragma unroll
    for (int cy = 0; cy < 2; ++cy) {
#pragma unroll
      for (int cx = 0; cx < 2; ++cx) {
        int xi = x0 + cx, yi = y0 + cy;
        if (xi >= 0 && xi < WF && yi >= 0 && yi < HF) {
          float wt = (cx ? fx : 1.0f - fx) * (cy ? fy : 1.0f - fy);
          float4 v = *(const float4*)(fb + ((size_t)yi * WF + xi) * CFE);
          vx += wt * v.x; vy += wt * v.y; vz += wt * v.z; vw += wt * v.w;
        }
      }
    }
    ax += w * vx; ay += w * vy; az += w * vz; aw += w * vw;
  }
  v4h o = { (_Float16)ax, (_Float16)ay, (_Float16)az, (_Float16)aw };
  *(v4h*)(agg16 + (size_t)m * CFE + c4) = o;
}

extern "C" void kernel_launch(void* const* d_in, const int* in_sizes, int n_in,
                              void* d_out, int out_size, void* d_ws, size_t ws_size,
                              hipStream_t stream) {
  const float* query     = (const float*)d_in[0];
  const float* qpos      = (const float*)d_in[1];
  const float* rho       = (const float*)d_in[2];
  const float* phi       = (const float*)d_in[3];
  const float* zz        = (const float*)d_in[4];
  const float* feats     = (const float*)d_in[5];
  const float* l2i       = (const float*)d_in[6];
  const float* intervals = (const float*)d_in[7];
  const float* pcr       = (const float*)d_in[8];
  const float* imsz      = (const float*)d_in[9];
  const float* Wa        = (const float*)d_in[10];
  const float* ba        = (const float*)d_in[11];
  const float* Wo        = (const float*)d_in[12];
  const float* bo        = (const float*)d_in[13];
  float* outp = (float*)d_out;

  const int nq = in_sizes[2];           // 16384 (128x128 BEV)
  const int bs = in_sizes[4] / nq;      // 2
  const int M  = bs * nq;               // 32768 rows

  // workspace carve-out (256B aligned)
  char* wsb = (char*)d_ws;
  size_t off = 0;
  auto carve = [&](size_t bytes) -> void* {
    void* p = wsb + off;
    off = (off + bytes + 255) & ~(size_t)255;
    return p;
  };
  _Float16* q16    = (_Float16*)carve((size_t)M * HID * sizeof(_Float16));
  _Float16* wa16   = (_Float16*)carve((size_t)KDIM * 32 * sizeof(_Float16));
  _Float16* wo16   = (_Float16*)carve((size_t)KDIM * HID * sizeof(_Float16));
  float*    awbuf  = (float*)   carve((size_t)M * 32 * sizeof(float));
  float*    wbuf   = (float*)   carve((size_t)M * N_CAM * sizeof(float));
  float*    ixy    = (float*)   carve((size_t)M * N_CAM * 2 * sizeof(float));
  float*    featsT = (float*)   carve((size_t)bs * N_CAM * HF * WF * CFE * sizeof(float));
  _Float16* agg16  = (_Float16*)carve((size_t)M * CFE * sizeof(_Float16));

  float* mask_out = outp + (size_t)M * HID;   // second tuple output, [bs,1,nq,N,1]

  const int TB = 256;

  // 1) q16 = f16(query + query_pos), (b,q) row order, float4 vectorized
  {
    dim3 g((nq * HID / 4 + TB - 1) / TB, bs);
    k_prep_q16<<<g, TB, 0, stream>>>(query, qpos, q16, nq, bs);
  }
  // 2) Wa/Wo -> fragment-native swizzled f16 (Wa zero-padded 24 -> 32 cols)
  k_prep_bswz<<<(32 * KDIM + TB - 1) / TB, TB, 0, stream>>>(Wa, wa16, 32, N_CAM * N_PTS);
  k_prep_bswz<<<(HID * KDIM + TB - 1) / TB, TB, 0, stream>>>(Wo, wo16, HID, HID);
  // 3) feats channel-last transpose
  {
    int total = bs * N_CAM * HF * WF * CFE;
    k_feats_T<<<(total + TB - 1) / TB, TB, 0, stream>>>(feats, featsT, total);
  }
  // 4) WMMA GEMM1: aw[M,32] = q16[M,256] @ Wa
  {
    int tiles = (M / 16) * (32 / 16);
    k_wmma_gemm<0><<<(tiles + 7) / 8, TB, 0, stream>>>(q16, wa16, awbuf, M, 32,
                                                       nullptr, nullptr, nullptr, nq, bs);
  }
  // 5) geometry + mask + sigmoid weights (also writes mask tuple output)
  {
    dim3 g((nq + TB - 1) / TB, bs);
    k_geom<<<g, TB, 0, stream>>>(rho, phi, zz, l2i, intervals, pcr, imsz,
                                 awbuf, ba, wbuf, ixy, mask_out, nq);
  }
  // 6) bilinear sample + camera-weighted combine -> agg16[M,256]
  {
    dim3 g(nq, bs);
    k_sample<<<g, 64, 0, stream>>>(featsT, wbuf, ixy, agg16, nq);
  }
  // 7) WMMA GEMM2 with fused epilogue: out[(q,b),c] = agg @ Wo + bo + query
  {
    int tiles = (M / 16) * (HID / 16);
    k_wmma_gemm<1><<<(tiles + 7) / 8, TB, 0, stream>>>(agg16, wo16, nullptr, M, HID,
                                                       bo, query, outp, nq, bs);
  }
}
